// Relation_26070451486904
// MI455X (gfx1250) — matrix operand order repaired and checked by reading, not compiled, not run on hardware
//
#include <hip/hip_runtime.h>
#include <hip/hip_bf16.h>

typedef __attribute__((ext_vector_type(16))) __bf16 v16bf;
typedef __attribute__((ext_vector_type(8)))  float  v8f;

#define NB 4
#define NT 512      // N tokens
#define HH 512      // hidden
#define DR 64

// ---------------- Pass 1: per-token projections ----------------
// q = rope(node@Wq.T+bq)  -> qg (f32, [B*N, 64])
// k = rope(node@Wk.T+bk)  -> kA (bf16, WMMA-A-fragment layout [B][32 mtiles][2 kc][32 lanes][16 e])
// p2 = (node@Wp.T+bp)@Wt.T -> p2 (f32, [B*N, 64])
// v = node@Wv.T+bv        -> vg (f32, [B*N, 512])
__global__ __launch_bounds__(256)
void rel_proj_kernel(const float* __restrict__ node, const float* __restrict__ node_mass,
                     const float* __restrict__ Wq, const float* __restrict__ bq,
                     const float* __restrict__ Wk, const float* __restrict__ bk,
                     const float* __restrict__ Wp, const float* __restrict__ bp,
                     const float* __restrict__ Wv, const float* __restrict__ bv,
                     const float* __restrict__ Wt,
                     float* __restrict__ qg, __bf16* __restrict__ kA,
                     float* __restrict__ p2, float* __restrict__ vg)
{
    __shared__ float nrow[HH];
    __shared__ float lin[3 * DR];   // qlin, klin, plin
    __shared__ float kr[DR];

    const int tid = threadIdx.x;
    const int t   = blockIdx.x;        // b*512 + tok
    const int b   = t >> 9;
    const int tok = t & 511;

    nrow[tid]       = node[(size_t)t * HH + tid];
    nrow[tid + 256] = node[(size_t)t * HH + tid + 256];
    __syncthreads();

    if (tid < 192) {
        const int which = tid >> 6;     // 0=q,1=k,2=p
        const int o     = tid & 63;
        const float* W  = (which == 0) ? Wq : ((which == 1) ? Wk : Wp);
        const float* bb = (which == 0) ? bq : ((which == 1) ? bk : bp);
        const float* wr = W + (size_t)o * HH;
        float acc = bb[o];
        #pragma unroll 8
        for (int h = 0; h < HH; ++h) acc += nrow[h] * wr[h];
        lin[which * DR + o] = acc;
    }

    // v projection: every thread computes 2 outputs
    {
        float v0 = bv[tid], v1 = bv[tid + 256];
        const float* wr0 = Wv + (size_t)tid * HH;
        const float* wr1 = Wv + (size_t)(tid + 256) * HH;
        #pragma unroll 8
        for (int h = 0; h < HH; ++h) { v0 += nrow[h] * wr0[h]; v1 += nrow[h] * wr1[h]; }
        vg[(size_t)t * HH + tid]       = v0;
        vg[(size_t)t * HH + tid + 256] = v1;
    }
    __syncthreads();

    // RoPE for q (tid 0..31) and k (tid 32..63)
    if (tid < 64) {
        const int which = tid >> 5;
        const int l     = tid & 31;
        const float s = node_mass[(size_t)t * DR + l];
        const float c = node_mass[(size_t)t * DR + 32 + l];
        const float* src = lin + which * DR;
        const float x0 = src[2 * l], x1 = src[2 * l + 1];
        const float r0 = x0 * c - x1 * s;
        const float r1 = x1 * c + x0 * s;
        if (which == 0) {
            qg[(size_t)t * DR + l]      = r0;
            qg[(size_t)t * DR + 32 + l] = r1;
        } else {
            kr[l] = r0; kr[32 + l] = r1;
        }
    }
    __syncthreads();

    // scatter k into WMMA A-fragment layout (bf16)
    if (tid < 64) {
        const int e    = tid & 15;
        const int half = (tid >> 4) & 1;
        const int kc   = tid >> 5;
        const int K    = 32 * kc + 16 * half + e;
        const int mt   = tok >> 4;
        const int ml   = tok & 15;
        kA[(((size_t)(b * 32 + mt) * 2 + kc) * 32 + (half * 16 + ml)) * 16 + e] = (__bf16)kr[K];
    }

    // p2 = p @ Wt.T
    if (tid < 64) {
        const float* pr = lin + 2 * DR;
        const float* wr = Wt + (size_t)tid * DR;
        float acc = 0.f;
        #pragma unroll
        for (int i = 0; i < DR; ++i) acc += pr[i] * wr[i];
        p2[(size_t)t * DR + tid] = acc;
    }
}

// ---------------- Pass 1b: small precomputes ----------------
__global__ __launch_bounds__(64)
void rel_prep_emb(const float* __restrict__ de, const float* __restrict__ pred_pad,
                  const float* __restrict__ Wt, float* __restrict__ de2, float* __restrict__ pp2)
{
    const int d  = blockIdx.x;     // 0..99 = dist_emb rows, 100 = pred_pad
    const int j  = threadIdx.x;
    const float* src = (d < 100) ? (de + (size_t)d * DR) : pred_pad;
    const float* wr  = Wt + (size_t)j * DR;
    float acc = 0.f;
    #pragma unroll
    for (int i = 0; i < DR; ++i) acc += src[i] * wr[i];
    if (d < 100) de2[(size_t)d * DR + j] = acc; else pp2[j] = acc;
}

__global__ __launch_bounds__(256)
void rel_transpose_wo(const float* __restrict__ Wo, float* __restrict__ WoT)
{
    const int idx = blockIdx.x * 256 + threadIdx.x;   // 0 .. 512*512-1
    const int h  = idx >> 9;
    const int hp = idx & 511;
    WoT[(size_t)hp * HH + h] = Wo[idx];
}

// ---------------- Pass 2: fused relation-attention per (b,n) ----------------
__global__ __launch_bounds__(256, 4)
void rel_main_kernel(const float* __restrict__ node,
                     const int* __restrict__ dist,
                     const int* __restrict__ preds,
                     const unsigned char* __restrict__ mask,
                     const float* __restrict__ Wt,
                     const float* __restrict__ bo,
                     const float* __restrict__ ln_g, const float* __restrict__ ln_b,
                     const float* __restrict__ qg,
                     const __bf16* __restrict__ kA,
                     const float* __restrict__ p2,
                     const float* __restrict__ vg,
                     const float* __restrict__ de2,
                     const float* __restrict__ pp2,
                     const float* __restrict__ WoT,
                     float* __restrict__ out)
{
    __shared__ float relS[NT * DR];      // 128 KB
    __shared__ float qs[DR];
    __shared__ float cred[4 * DR];
    __shared__ float cfin[DR];
    __shared__ float csum[4 * DR];
    __shared__ float csfin[DR];
    __shared__ float ppart[4 * DR * 8];  // 8 KB
    __shared__ float postS[HH];
    __shared__ float red1[256];
    __shared__ float red2[256];

    const int tid  = threadIdx.x;
    const int bn   = blockIdx.x;         // b*512 + n
    const int b    = bn >> 9;
    const int lane = tid & 31;
    const int w    = tid >> 5;

    if (tid < DR) qs[tid] = qg[(size_t)bn * DR + tid];
    __syncthreads();

    // Build B' fragments: B'[i,j] = Wt[j,i] * q[i]   (64x64, bf16)
    v16bf bfrag[2][4];
    {
        const int jj = lane & 15;
        const int hk = (lane >> 4) * 16;
        #pragma unroll
        for (int kc = 0; kc < 2; ++kc) {
            #pragma unroll
            for (int cg = 0; cg < 4; ++cg) {
                const float* wrow = Wt + (size_t)(cg * 16 + jj) * DR + kc * 32 + hk;
                const float* qrow = qs + kc * 32 + hk;
                v16bf f;
                #pragma unroll
                for (int e = 0; e < 16; ++e) f[e] = (__bf16)(wrow[e] * qrow[e]);
                bfrag[kc][cg] = f;
            }
        }
    }

    // rel = k @ B' via WMMA bf16 -> f32; 4 m-tiles of 16 rows per wave.
    // Keep this loop rolled: unrolling quadruples live accumulators (VGPR blowup
    // past 256 observed) with no reuse benefit — B fragments already carry it.
    #pragma unroll 1
    for (int t4 = 0; t4 < 4; ++t4) {
        const int mt = w + t4 * 8;
        const v16bf a0 = *(const v16bf*)(kA + (((size_t)(b * 32 + mt) * 2 + 0) * 32 + lane) * 16);
        const v16bf a1 = *(const v16bf*)(kA + (((size_t)(b * 32 + mt) * 2 + 1) * 32 + lane) * 16);
        #pragma unroll
        for (int cg = 0; cg < 4; ++cg) {
            v8f acc = {};
            acc = __builtin_amdgcn_wmma_f32_16x16x32_bf16(false, a0, false, bfrag[0][cg],
                                                          (short)0, acc, false, false);
            acc = __builtin_amdgcn_wmma_f32_16x16x32_bf16(false, a1, false, bfrag[1][cg],
                                                          (short)0, acc, false, false);
            const int jj = cg * 16 + (lane & 15);
            const int mb = mt * 16 + ((lane >> 4) << 3);
            #pragma unroll
            for (int r = 0; r < 8; ++r)
                relS[(mb + r) * DR + jj] = acc[r];
        }
    }
    __syncthreads();

    // Gathered additive terms + mask:  rel += dist_emb2[dist] + (idx==0 ? pred_pad2 : p2[idx])
    const int j  = tid & 63;
    const int q4 = tid >> 6;
    const size_t base_nm = (size_t)bn * NT;
    for (int mm = 0; mm < 128; ++mm) {
        const int m = q4 * 128 + mm;
        const size_t nm = base_nm + m;
        const int d  = dist[nm];
        const int pi = preds[nm];
        float add = de2[(size_t)d * DR + j];
        add += (pi == 0) ? pp2[j] : p2[((size_t)b * NT + pi) * DR + j];
        float val = relS[m * DR + j] + add;
        if (!mask[nm]) val = -__builtin_inff();
        relS[m * DR + j] = val;
    }
    __syncthreads();

    // Softmax over m per column j (normalization folded into the v-reduction)
    float lmax = -__builtin_inff();
    for (int mm = 0; mm < 128; ++mm)
        lmax = fmaxf(lmax, relS[(q4 * 128 + mm) * DR + j]);
    cred[q4 * DR + j] = lmax;
    __syncthreads();
    if (tid < DR)
        cfin[tid] = fmaxf(fmaxf(cred[tid], cred[DR + tid]),
                          fmaxf(cred[2 * DR + tid], cred[3 * DR + tid]));
    __syncthreads();
    const float cm = cfin[j];
    float lsum = 0.f;
    for (int mm = 0; mm < 128; ++mm) {
        const int off = (q4 * 128 + mm) * DR + j;
        const float e = __expf(relS[off] - cm);
        relS[off] = e;
        lsum += e;
    }
    csum[q4 * DR + j] = lsum;
    __syncthreads();
    if (tid < DR)
        csfin[tid] = csum[tid] + csum[DR + tid] + csum[2 * DR + tid] + csum[3 * DR + tid];
    __syncthreads();

    // post[i, 0..7] = sum_m rel[m,i] * v[m, i*8 + 0..7]  (i == j for this thread)
    {
        float a0=0,a1=0,a2=0,a3=0,a4=0,a5=0,a6=0,a7=0;
        const float* vb = vg + (size_t)b * NT * HH;
        for (int mm = 0; mm < 128; ++mm) {
            const int m = q4 * 128 + mm;
            const float wgt = relS[m * DR + j];
            const float4* vp = (const float4*)(vb + (size_t)m * HH + j * 8);
            const float4 v0 = vp[0], v1 = vp[1];
            a0 += wgt * v0.x; a1 += wgt * v0.y; a2 += wgt * v0.z; a3 += wgt * v0.w;
            a4 += wgt * v1.x; a5 += wgt * v1.y; a6 += wgt * v1.z; a7 += wgt * v1.w;
        }
        float* pp = ppart + (q4 * DR + j) * 8;
        pp[0]=a0; pp[1]=a1; pp[2]=a2; pp[3]=a3; pp[4]=a4; pp[5]=a5; pp[6]=a6; pp[7]=a7;
    }
    __syncthreads();
    #pragma unroll
    for (int s = 0; s < 2; ++s) {
        const int idx = tid * 2 + s;        // 0..511  (= i*8 + j2)
        const int i2  = idx >> 3;
        const int j2  = idx & 7;
        const float sum = ppart[(0 * DR + i2) * 8 + j2] + ppart[(1 * DR + i2) * 8 + j2]
                        + ppart[(2 * DR + i2) * 8 + j2] + ppart[(3 * DR + i2) * 8 + j2];
        postS[idx] = sum / csfin[i2];
    }
    __syncthreads();

    // out = LN(node + post @ Wo.T + bo)
    float x0 = bo[tid]       + node[(size_t)bn * HH + tid];
    float x1 = bo[tid + 256] + node[(size_t)bn * HH + tid + 256];
    #pragma unroll 4
    for (int hp = 0; hp < HH; ++hp) {
        const float pv = postS[hp];
        const float* wr = WoT + (size_t)hp * HH;
        x0 += pv * wr[tid];
        x1 += pv * wr[tid + 256];
    }
    red1[tid] = x0 + x1;
    red2[tid] = x0 * x0 + x1 * x1;
    __syncthreads();
    for (int s = 128; s > 0; s >>= 1) {
        if (tid < s) { red1[tid] += red1[tid + s]; red2[tid] += red2[tid + s]; }
        __syncthreads();
    }
    const float mu  = red1[0] * (1.f / 512.f);
    const float var = red2[0] * (1.f / 512.f) - mu * mu;
    const float rs  = rsqrtf(var + 1e-5f);
    out[(size_t)bn * HH + tid]       = (x0 - mu) * rs * ln_g[tid]       + ln_b[tid];
    out[(size_t)bn * HH + tid + 256] = (x1 - mu) * rs * ln_g[tid + 256] + ln_b[tid + 256];
}

// ---------------- Launcher ----------------
extern "C" void kernel_launch(void* const* d_in, const int* in_sizes, int n_in,
                              void* d_out, int out_size, void* d_ws, size_t ws_size,
                              hipStream_t stream) {
    const float* node      = (const float*)d_in[0];
    const float* node_mass = (const float*)d_in[1];
    const int*   dist      = (const int*)  d_in[2];
    const int*   preds     = (const int*)  d_in[3];
    const unsigned char* rel_mask = (const unsigned char*)d_in[4];
    const float* Wq = (const float*)d_in[5];
    const float* bq = (const float*)d_in[6];
    const float* Wk = (const float*)d_in[7];
    const float* bk = (const float*)d_in[8];
    const float* Wp = (const float*)d_in[9];
    const float* bp = (const float*)d_in[10];
    const float* Wv = (const float*)d_in[11];
    const float* bv = (const float*)d_in[12];
    const float* dist_emb = (const float*)d_in[13];
    const float* Wt = (const float*)d_in[14];
    const float* Wo = (const float*)d_in[15];
    const float* bo = (const float*)d_in[16];
    const float* pred_pad = (const float*)d_in[17];
    const float* ln_g = (const float*)d_in[18];
    const float* ln_b = (const float*)d_in[19];
    float* out = (float*)d_out;

    char* ws = (char*)d_ws;
    float*  qg  = (float*)(ws + 0);                 // 512 KB
    float*  p2  = (float*)(ws + 524288);            // 512 KB
    float*  vg  = (float*)(ws + 1048576);           // 4 MB
    __bf16* kA  = (__bf16*)(ws + 5242880);          // 256 KB
    float*  de2 = (float*)(ws + 5505024);           // 25.6 KB
    float*  pp2 = (float*)(ws + 5530624);           // 256 B
    float*  WoT = (float*)(ws + 5530880);           // 1 MB

    rel_proj_kernel<<<NB * NT, 256, 0, stream>>>(node, node_mass, Wq, bq, Wk, bk, Wp, bp,
                                                 Wv, bv, Wt, qg, kA, p2, vg);
    rel_prep_emb<<<101, 64, 0, stream>>>(dist_emb, pred_pad, Wt, de2, pp2);
    rel_transpose_wo<<<(HH * HH) / 256, 256, 0, stream>>>(Wo, WoT);
    rel_main_kernel<<<NB * NT, 256, 0, stream>>>(node, dist, preds, rel_mask, Wt, bo,
                                                 ln_g, ln_b, qg, kA, p2, vg, de2, pp2,
                                                 WoT, out);
}
